// plnresCtdetLoss_35278861369826
// MI455X (gfx1250) — matrix operand order repaired and checked by reading, not compiled, not run on hardware
//
#include <hip/hip_runtime.h>
#include <hip/hip_bf16.h>

typedef __attribute__((ext_vector_type(16))) _Float16 v16h;
typedef __attribute__((ext_vector_type(8)))  float    v8f;

#define BB   32
#define RR   32
#define CC   32
#define NBOX 2
#define NCLS 20
#define CHL  174            // 6 + 2*(R+C) + 2*NC
#define CHP  348            // NB * CHL
#define CBASE 134           // 6 + 2*(R+C)
#define CELLS (BB*RR*CC)    // 32768
#define WAVES_PER_BLK 8

// ---------------------------------------------------------------------------
// Zero-fill d_out (harness poisons it; scatter only writes 1.0s).
// ---------------------------------------------------------------------------
__global__ void convbox_zero_kernel(float4* __restrict__ out, int n4) {
    int i = blockIdx.x * blockDim.x + threadIdx.x;
    int stride = gridDim.x * blockDim.x;
    float4 z = {0.f, 0.f, 0.f, 0.f};
    for (; i < n4; i += stride) out[i] = z;
}

// ---------------------------------------------------------------------------
// Wave-cooperative squared-norm of 32 per-lane 20-vectors via one Gram-matrix
// WMMA per 16-cell half: D = A * A^T (f16 in, f32 out), take the diagonal.
//   A-frag (16x32 f16): lane l -> row (l&15); halves 0-7 = K {0..7|8..15},
//                       halves 8-15 = K {16..23|24..31} (low/high half-wave).
//   B-frag (32x16 f16): lane l -> col (l&15); same K order as A-frag.
//   C/D (16x16 f32):    VGPR v, lanes 0-15 -> (M=v, N=lane),
//                       lanes 16-31 -> (M=v+8, N=lane-16).
// Must be called in uniform control flow (EXEC all ones).
// ---------------------------------------------------------------------------
__device__ __forceinline__ float wave_norm20_wmma(const float d[20],
                                                  _Float16 (*hb)[32],
                                                  float (*db)[32],
                                                  int lane) {
    // stage this lane's diff vector (padded to K=32) into LDS
    #pragma unroll
    for (int k = 0; k < 20; ++k) hb[lane][k] = (_Float16)d[k];
    #pragma unroll
    for (int k = 20; k < 32; ++k) hb[lane][k] = (_Float16)0.0f;
    __syncthreads();

    const int  m   = lane & 15;
    const bool lo  = (lane < 16);
    const int  klo = lo ? 0 : 8;
    const int  khi = lo ? 16 : 24;
    const int  kb  = lo ? 0 : 16;

    #pragma unroll
    for (int g = 0; g < 2; ++g) {
        const _Float16* row = hb[g * 16 + m];   // A row == B column (Gram)
        v16h a, bfrag;
        #pragma unroll
        for (int j = 0; j < 8; ++j) {
            a[j]     = row[klo + j];
            a[8 + j] = row[khi + j];
        }
        #pragma unroll
        for (int j = 0; j < 8; ++j) { bfrag[j] = row[klo + j]; bfrag[8 + j] = row[khi + j]; }

        v8f acc = {};
        acc = __builtin_amdgcn_wmma_f32_16x16x32_f16(
            /*neg_a=*/false, a, /*neg_b=*/false, bfrag,
            /*c_mod=*/(short)0, acc, /*reuse_a=*/false, /*reuse_b=*/false);

        #pragma unroll
        for (int v = 0; v < 8; ++v) db[g * 8 + v][lane] = acc[v];
    }
    __syncthreads();

    // diagonal element for this lane's cell
    const int g  = lane >> 4;          // which 16-cell group
    const int mm = lane & 15;          // M == N
    const int v  = mm & 7;
    const int dl = (mm < 8) ? mm : (mm + 16);
    return db[g * 8 + v][dl];
}

// ---------------------------------------------------------------------------
// One thread per cell; one wave per 32 cells (for the WMMA cat term).
// ---------------------------------------------------------------------------
__global__ __launch_bounds__(256) void convbox_match_kernel(
        const float* __restrict__ pred,
        const float* __restrict__ label,
        float* __restrict__ out) {
    __shared__ _Float16 s_h[WAVES_PER_BLK][32][32];
    __shared__ float    s_d[WAVES_PER_BLK][16][32];

    const int tid  = blockIdx.x * blockDim.x + threadIdx.x;   // cell id
    const int wv   = threadIdx.x >> 5;
    const int lane = threadIdx.x & 31;

    const int c = tid & 31;
    const int r = (tid >> 5) & 31;
    const int b = tid >> 10;

    const float* lbl = label + (size_t)tid * CHL;
    const float* prd = pred  + (size_t)tid * CHP;

    // first-max argmax over the one-hot corner pointers
    int rowc = 0; float bv = lbl[6];
    for (int k = 1; k < 32; ++k) { float v = lbl[6 + k];  if (v > bv) { bv = v; rowc = k; } }
    int colc = 0; bv = lbl[38];
    for (int k = 1; k < 32; ++k) { float v = lbl[38 + k]; if (v > bv) { bv = v; colc = k; } }

    const int ccell = (b * 32 + rowc) * 32 + colc;
    const float* Lc = label + (size_t)ccell * CHL;
    const float* Pc = pred  + (size_t)ccell * CHP;

    const float inv = 1.0f / 32.0f;
    const float xm_t = (lbl[2] + (float)c) * inv;
    const float ym_t = (lbl[3] + (float)r) * inv;
    const float xc_t = (Lc[4] + (float)colc) * inv;
    const float yc_t = (Lc[5] + (float)rowc) * inv;
    const float tw  = fabsf(xm_t - xc_t) * 2.0f;
    const float th_ = fabsf(ym_t - yc_t) * 2.0f;

    // cat[i] = ||pred_main - label_main||^2 + ||Pc_corner - Lc_corner||^2 (WMMA)
    float cat[2];
    #pragma unroll
    for (int i = 0; i < 2; ++i) {
        float d[20];
        #pragma unroll
        for (int k = 0; k < 20; ++k) d[k] = prd[268 + i * 20 + k] - lbl[134 + k];
        float cm = wave_norm20_wmma(d, s_h[wv], s_d[wv], lane);
        #pragma unroll
        for (int k = 0; k < 20; ++k) d[k] = Pc[308 + i * 20 + k] - Lc[154 + k];
        float cc2 = wave_norm20_wmma(d, s_h[wv], s_d[wv], lane);
        cat[i] = cm + cc2;
    }

    float score[2];
    #pragma unroll
    for (int i = 0; i < 2; ++i) {
        const float conn_m = prd[12 + i * 64 + rowc] * prd[12 + i * 64 + 32 + colc];
        const float conn_c = Pc[140 + i * 64 + r]    * Pc[140 + i * 64 + 32 + c];
        const float conn   = 0.5f * (conn_m + conn_c);

        const float xm_p = prd[4 + 2 * i + c]    * inv;
        const float ym_p = prd[5 + 2 * i + r]    * inv;
        const float xc_p = Pc[8 + 2 * i + colc]  * inv;
        const float yc_p = Pc[9 + 2 * i + rowc]  * inv;
        const float ow = fabsf(xm_p - xc_p) * 2.0f;
        const float oh = fabsf(ym_p - yc_p) * 2.0f;

        const float w = fminf(xm_p + ow * 0.5f, xm_t + tw  * 0.5f)
                      - fmaxf(xm_p - ow * 0.5f, xm_t - tw  * 0.5f);
        const float h = fminf(ym_p + oh * 0.5f, ym_t + th_ * 0.5f)
                      - fmaxf(ym_p - oh * 0.5f, ym_t - th_ * 0.5f);
        const float inter = ((w < 0.0f) || (h < 0.0f)) ? 0.0f : w * h;
        const float uni   = ow * oh + tw * th_ - inter;
        const float iou   = inter / uni;
        const float rmse  = (xm_p - xm_t) * (xm_p - xm_t) + (ym_p - ym_t) * (ym_p - ym_t)
                          + (ow - tw) * (ow - tw) + (oh - th_) * (oh - th_);
        score[i] = conn * (iou - rmse + 0.1f) + 0.1f * (2.0f - cat[i]);
    }
    const int best = (score[1] > score[0]) ? 1 : 0;

    // scatter 1.0s at flagged cells (idempotent -> plain stores, no atomics)
    if (lbl[0] == 1.0f) {
        float* outP  = out;
        float* outL1 = out + (size_t)CELLS * CHP;
        float* outL2 = outL1 + (size_t)CELLS * CHL;
        #pragma unroll
        for (int n = 0; n < 2; ++n) {
            const int rt = n ? rowc : r;
            const int ct = n ? colc : c;
            const int cellT = (b * 32 + rt) * 32 + ct;
            float* pm = outP  + (size_t)cellT * CHP;
            float* l1 = outL1 + (size_t)cellT * CHL;
            float* l2 = outL2 + (size_t)cellT * CHL;

            // pred_mask channels
            pm[n * 2 + best] = 1.0f;
            const int c2 = ((1 + n) * 2 + best) * 2;
            pm[c2] = 1.0f; pm[c2 + 1] = 1.0f;
            const int c3 = (6 + n * 64) * 2 + best * 64;
            for (int k = 0; k < 64; ++k) pm[c3 + k] = 1.0f;
            const int c4 = (134 + n * 20) * 2 + best * 20;
            for (int k = 0; k < 20; ++k) pm[c4 + k] = 1.0f;

            // label_mask channels (written to both output copies)
            l1[n] = 1.0f;                l2[n] = 1.0f;
            l1[2 * (1 + n)]     = 1.0f;  l2[2 * (1 + n)]     = 1.0f;
            l1[2 * (1 + n) + 1] = 1.0f;  l2[2 * (1 + n) + 1] = 1.0f;
            for (int k = 0; k < 64; ++k) { l1[6 + n * 64 + k]   = 1.0f; l2[6 + n * 64 + k]   = 1.0f; }
            for (int k = 0; k < 20; ++k) { l1[134 + n * 20 + k] = 1.0f; l2[134 + n * 20 + k] = 1.0f; }
        }
    }
}

extern "C" void kernel_launch(void* const* d_in, const int* in_sizes, int n_in,
                              void* d_out, int out_size, void* d_ws, size_t ws_size,
                              hipStream_t stream) {
    (void)in_sizes; (void)n_in; (void)d_ws; (void)ws_size;
    const float* pred  = (const float*)d_in[0];
    const float* label = (const float*)d_in[1];
    float* out = (float*)d_out;

    // out_size = 32768 * (348 + 174 + 174) = 22,806,528 floats (divisible by 4)
    const int n4 = out_size / 4;
    convbox_zero_kernel<<<1024, 256, 0, stream>>>((float4*)out, n4);
    convbox_match_kernel<<<CELLS / 256, 256, 0, stream>>>(pred, label, out);
}
